// BertGeneratorModel_66279935312510
// MI455X (gfx1250) — compile-verified, matrix operand order checked by reading, not run on hardware
//
#include <hip/hip_runtime.h>

// ---------------------------------------------------------------------------
// BERT forward for MI455X (gfx1250, wave32, WMMA).
// All heavy math runs on v_wmma_f32_16x16x32_bf16 with f32 accumulation.
// GEMM A-tiles are staged with GLOBAL_LOAD_ASYNC_TO_LDS (ASYNCcnt path).
// ---------------------------------------------------------------------------

#define B_   8
#define S_   512
#define H_   768
#define L_   12
#define NH_  12
#define DH_  64
#define FF_  3072
#define V_   21128
#define M_   (B_ * S_)          // 4096 token rows
#define PAD_ID 0
#define SEP_ID 102

#define AS1 __attribute__((address_space(1)))
#define AS3 __attribute__((address_space(3)))

#if defined(__has_builtin)
#if __has_builtin(__builtin_amdgcn_global_load_async_to_lds_b128)
#define ASYNC_LDS 1
#endif
#endif

typedef __attribute__((ext_vector_type(16))) __bf16 v16bf;
typedef __attribute__((ext_vector_type(8)))  float  v8f;
typedef int v4i_ __attribute__((vector_size(16)));   // matches builtin proto
typedef unsigned short ushort_t;
typedef unsigned int   uint_t;

union Frag16 {                      // one 16x32 bf16 WMMA operand (per lane)
    v16bf    v;
    uint_t   u[8];
    ushort_t s[16];
};

// fp32 -> bf16 (round-to-nearest-even), bit level only
__device__ __forceinline__ ushort_t f2bf(float f) {
    uint_t u = __float_as_uint(f);
    uint_t r = ((u >> 16) & 1u) + 0x7FFFu;
    return (ushort_t)((u + r) >> 16);
}
__device__ __forceinline__ float bf2f(ushort_t s) {
    return __uint_as_float(((uint_t)s) << 16);
}

// K-offset of dword pair j (j=0..7) inside a 16-bit 16x32 A/B operand,
// per CDNA5 ISA 7.12.2 (half = lane>>4).
__device__ __forceinline__ int kkof(int j, int half) {
    return 2 * j + 8 * half + (j >= 4 ? 8 : 0);
}

__device__ __forceinline__ v8f wmma_bf16(v16bf a, v16bf b, v8f c) {
    return __builtin_amdgcn_wmma_f32_16x16x32_bf16(
        false, a, false, b, (short)0, c, false, false);
}

#ifdef ASYNC_LDS
__device__ __forceinline__ void async_copy_b128(const void* gsrc, void* ldst) {
    __builtin_amdgcn_global_load_async_to_lds_b128(
        (AS1 v4i_*)(unsigned long long)gsrc,
        (AS3 v4i_*)(unsigned int)(unsigned long long)ldst, 0, 0);
}
__device__ __forceinline__ void wait_async0() {
#if __has_builtin(__builtin_amdgcn_s_wait_asynccnt)
    __builtin_amdgcn_s_wait_asynccnt(0);
#else
    asm volatile("s_wait_asynccnt 0" ::: "memory");
#endif
}
#endif

__device__ __forceinline__ float rowmax16(float v) {
    #pragma unroll
    for (int o = 1; o < 16; o <<= 1) v = fmaxf(v, __shfl_xor(v, o));
    return v;
}
__device__ __forceinline__ float rowsum16(float v) {
    #pragma unroll
    for (int o = 1; o < 16; o <<= 1) v += __shfl_xor(v, o);
    return v;
}

// ---------------------------------------------------------------------------
// fp32 -> bf16 weight conversion (grid-stride)
// ---------------------------------------------------------------------------
__global__ __launch_bounds__(256) void cvt_bf16_kernel(
        const float* __restrict__ src, ushort_t* __restrict__ dst, size_t n) {
    size_t i = (size_t)blockIdx.x * blockDim.x + threadIdx.x;
    size_t stride = (size_t)gridDim.x * blockDim.x;
    for (; i < n; i += stride) dst[i] = f2bf(src[i]);
}

// ---------------------------------------------------------------------------
// qlen[b] = argmax(ids==SEP) + 1  (first SEP, 0 if none -> 1)
// ---------------------------------------------------------------------------
__global__ __launch_bounds__(32) void qlen_kernel(
        const int* __restrict__ ids, int* __restrict__ qlen) {
    int b = blockIdx.x, lane = threadIdx.x;
    int best = S_;
    for (int j = lane; j < S_; j += 32)
        if (ids[b * S_ + j] == SEP_ID) best = min(best, j);
    #pragma unroll
    for (int o = 1; o < 32; o <<= 1) best = min(best, __shfl_xor(best, o));
    if (lane == 0) qlen[b] = (best == S_ ? 0 : best) + 1;
}

// ---------------------------------------------------------------------------
// Embedding gather + LayerNorm.  One block per token row (H=768, 256 thr x 3).
// ---------------------------------------------------------------------------
__global__ __launch_bounds__(256) void embed_ln_kernel(
        const int*   __restrict__ ids,
        const float* __restrict__ wemb, const float* __restrict__ pemb,
        const float* __restrict__ temb,
        const float* __restrict__ g, const float* __restrict__ bta,
        float* __restrict__ hF, ushort_t* __restrict__ hB) {
    __shared__ float redS[8], redQ[8], stat[2];
    int row = blockIdx.x, s = row % S_;
    int id  = ids[row];
    float x[3], s1 = 0.f, s2 = 0.f;
    #pragma unroll
    for (int t = 0; t < 3; t++) {
        int c = threadIdx.x + t * 256;
        x[t] = wemb[(size_t)id * H_ + c] + pemb[(size_t)s * H_ + c] + temb[c];
        s1 += x[t]; s2 += x[t] * x[t];
    }
    #pragma unroll
    for (int o = 1; o < 32; o <<= 1) { s1 += __shfl_xor(s1, o); s2 += __shfl_xor(s2, o); }
    if ((threadIdx.x & 31) == 0) { redS[threadIdx.x >> 5] = s1; redQ[threadIdx.x >> 5] = s2; }
    __syncthreads();
    if (threadIdx.x == 0) {
        float a = 0.f, b2 = 0.f;
        #pragma unroll
        for (int i = 0; i < 8; i++) { a += redS[i]; b2 += redQ[i]; }
        float mu  = a / (float)H_;
        float var = b2 / (float)H_ - mu * mu;
        stat[0] = mu; stat[1] = rsqrtf(var + 1e-12f);
    }
    __syncthreads();
    float mu = stat[0], inv = stat[1];
    #pragma unroll
    for (int t = 0; t < 3; t++) {
        int c = threadIdx.x + t * 256;
        float y = (x[t] - mu) * inv * g[c] + bta[c];
        hF[(size_t)row * H_ + c] = y;
        hB[(size_t)row * H_ + c] = f2bf(y);
    }
}

// ---------------------------------------------------------------------------
// Residual + LayerNorm:  h = LN(h + t)
// ---------------------------------------------------------------------------
__global__ __launch_bounds__(256) void add_ln_kernel(
        float* __restrict__ hF, const float* __restrict__ tF,
        const float* __restrict__ g, const float* __restrict__ bta,
        ushort_t* __restrict__ hB) {
    __shared__ float redS[8], redQ[8], stat[2];
    int row = blockIdx.x;
    float x[3], s1 = 0.f, s2 = 0.f;
    #pragma unroll
    for (int t = 0; t < 3; t++) {
        int c = threadIdx.x + t * 256;
        x[t] = hF[(size_t)row * H_ + c] + tF[(size_t)row * H_ + c];
        s1 += x[t]; s2 += x[t] * x[t];
    }
    #pragma unroll
    for (int o = 1; o < 32; o <<= 1) { s1 += __shfl_xor(s1, o); s2 += __shfl_xor(s2, o); }
    if ((threadIdx.x & 31) == 0) { redS[threadIdx.x >> 5] = s1; redQ[threadIdx.x >> 5] = s2; }
    __syncthreads();
    if (threadIdx.x == 0) {
        float a = 0.f, b2 = 0.f;
        #pragma unroll
        for (int i = 0; i < 8; i++) { a += redS[i]; b2 += redQ[i]; }
        float mu  = a / (float)H_;
        float var = b2 / (float)H_ - mu * mu;
        stat[0] = mu; stat[1] = rsqrtf(var + 1e-12f);
    }
    __syncthreads();
    float mu = stat[0], inv = stat[1];
    #pragma unroll
    for (int t = 0; t < 3; t++) {
        int c = threadIdx.x + t * 256;
        float y = (x[t] - mu) * inv * g[c] + bta[c];
        hF[(size_t)row * H_ + c] = y;
        hB[(size_t)row * H_ + c] = f2bf(y);
    }
}

// ---------------------------------------------------------------------------
// Tiled WMMA GEMM:  C[M,N] = A[M,K](bf16) * W[K,N](bf16) + bias
// Block tile 128x128, 8 waves, each wave 64x32 (4x2 accumulators).
// A tile staged via GLOBAL_LOAD_ASYNC_TO_LDS_B128 (overlaps with B transpose).
// EPI: 0 = bias -> bf16 out; 1 = bias+GELU -> bf16 out; 2 = bias -> f32 out
// ---------------------------------------------------------------------------
template <int EPI>
__global__ __launch_bounds__(256) void gemm_bf16_kernel(
        const ushort_t* __restrict__ A, const ushort_t* __restrict__ Bw,
        const float* __restrict__ bias,
        float* __restrict__ outF, ushort_t* __restrict__ outB,
        int Mdim, int N, int K) {
    __shared__ __align__(16) uint_t   As_u[128 * 20];   // [row][k/2], stride 20 dwords
    __shared__ __align__(16) ushort_t Bs[128 * 40];     // [col][k],   stride 40 shorts

    const int tid  = threadIdx.x;
    const int bm   = blockIdx.y * 128;
    const int bn   = blockIdx.x * 128;
    const int wave = tid >> 5, lane = tid & 31;
    const int wm   = (wave >> 2) * 64;     // 2 wave rows
    const int wn   = (wave & 3) * 32;      // 4 wave cols
    const int half = lane >> 4, lr = lane & 15;

    const uint_t* Au = reinterpret_cast<const uint_t*>(A);
    (void)Au;

    v8f acc[4][2];
    #pragma unroll
    for (int i = 0; i < 4; i++)
        #pragma unroll
        for (int n = 0; n < 2; n++)
            #pragma unroll
            for (int e = 0; e < 8; e++) acc[i][n][e] = 0.f;

    for (int k0 = 0; k0 < K; k0 += 32) {
        // ---- stage A tile 128x32 (dword pairs, fragment-ready layout) ----
#ifdef ASYNC_LDS
        // async DMA global->LDS, no VGPR round trip; overlaps with B transpose
        #pragma unroll
        for (int v = tid; v < 512; v += 256) {
            int row = v >> 2, kp = (v & 3) * 8;
            async_copy_b128(A + (size_t)(bm + row) * K + k0 + kp,
                            &As_u[row * 20 + (kp >> 1)]);
        }
#else
        #pragma unroll
        for (int v = tid; v < 512; v += 256) {
            int row = v >> 2, kp = (v & 3) * 8;
            size_t gi = ((size_t)(bm + row) * K + k0 + kp) >> 1;
            int lb = row * 20 + (kp >> 1);
            #pragma unroll
            for (int i = 0; i < 4; i++) As_u[lb + i] = Au[gi + i];
        }
#endif
        // ---- stage B tile 32x128 transposed into [col][k] ----
        #pragma unroll
        for (int v = tid; v < 512; v += 256) {
            int kr = v >> 4, nc = (v & 15) * 8;
            int gc = bn + nc;
            const ushort_t* src = Bw + (size_t)(k0 + kr) * N + gc;
            ushort_t e[8];
            if (gc + 8 <= N) {
                const uint_t* s4 = reinterpret_cast<const uint_t*>(src);
                #pragma unroll
                for (int i = 0; i < 4; i++) {
                    uint_t t = s4[i];
                    e[2 * i]     = (ushort_t)(t & 0xFFFFu);
                    e[2 * i + 1] = (ushort_t)(t >> 16);
                }
            } else {
                #pragma unroll
                for (int i = 0; i < 8; i++) e[i] = (gc + i < N) ? src[i] : (ushort_t)0;
            }
            #pragma unroll
            for (int i = 0; i < 8; i++) Bs[(nc + i) * 40 + kr] = e[i];
        }
        // prefetch next K tile of A (global_prefetch_b8 path)
        if (k0 + 32 < K)
            __builtin_prefetch(A + (size_t)(bm + (tid >> 1)) * K + k0 + 32, 0, 1);
#ifdef ASYNC_LDS
        wait_async0();
#endif
        __syncthreads();

        // ---- fragments + 8 WMMA ----
        Frag16 af[4], bf[2];
        #pragma unroll
        for (int i = 0; i < 4; i++)
            #pragma unroll
            for (int j = 0; j < 8; j++)
                af[i].u[j] = As_u[(wm + 16 * i + lr) * 20 + (kkof(j, half) >> 1)];
        #pragma unroll
        for (int n = 0; n < 2; n++)
            #pragma unroll
            for (int j = 0; j < 8; j++)
                bf[n].u[j] = *reinterpret_cast<const uint_t*>(
                    &Bs[(wn + 16 * n + lr) * 40 + kkof(j, half)]);
        #pragma unroll
        for (int i = 0; i < 4; i++)
            #pragma unroll
            for (int n = 0; n < 2; n++)
                acc[i][n] = wmma_bf16(af[i].v, bf[n].v, acc[i][n]);
        __syncthreads();
    }

    // ---- epilogue ----
    #pragma unroll
    for (int i = 0; i < 4; i++)
        #pragma unroll
        for (int n = 0; n < 2; n++)
            #pragma unroll
            for (int vr = 0; vr < 8; vr++) {
                int row = bm + wm + 16 * i + vr + 8 * half;
                int col = bn + wn + 16 * n + lr;
                if (row < Mdim && col < N) {
                    float val = acc[i][n][vr] + bias[col];
                    if (EPI == 1)
                        val = 0.5f * val * (1.0f + erff(val * 0.70710678118654752f));
                    if (EPI == 2)
                        outF[(size_t)row * N + col] = val;
                    else
                        outB[(size_t)row * N + col] = f2bf(val);
                }
            }
}

// ---------------------------------------------------------------------------
// Flash attention: one wave per (b, head, 16-query tile).
// QK^T and P.V via WMMA; online softmax with 16-lane shuffle reductions;
// P transposed through LDS into A-fragment layout.
// ---------------------------------------------------------------------------
__global__ __launch_bounds__(32) void flash_attn_kernel(
        const ushort_t* __restrict__ qb, const ushort_t* __restrict__ kb,
        const ushort_t* __restrict__ vb, ushort_t* __restrict__ ctxb,
        const int* __restrict__ ids, const int* __restrict__ qlen) {
    __shared__ __align__(16) ushort_t Ps[16 * 40];   // P tile [qrow][key]

    const int blk = blockIdx.x;
    const int qt  = blk % (S_ / 16);
    const int hh  = (blk / (S_ / 16)) % NH_;
    const int b   = blk / ((S_ / 16) * NH_);
    const int q0  = qt * 16;
    const int lane = threadIdx.x, half = lane >> 4, lr = lane & 15;
    const int qlv = qlen[b];
    const float scale = 0.125f;   // 1/sqrt(64)

    const uint_t* qbu = reinterpret_cast<const uint_t*>(qb);
    const uint_t* kbu = reinterpret_cast<const uint_t*>(kb);

    // Q fragments over feature halves 0..31 / 32..63 (A-layout: row = lane&15)
    Frag16 qa[2];
    #pragma unroll
    for (int fi = 0; fi < 2; fi++)
        #pragma unroll
        for (int j = 0; j < 8; j++)
            qa[fi].u[j] = qbu[((size_t)(b * S_ + q0 + lr) * H_ +
                               hh * DH_ + 32 * fi + kkof(j, half)) >> 1];

    // per-lane row metadata (this lane's 8 C-rows)
    int   rowi[8];
    float padi[8];
    #pragma unroll
    for (int vr = 0; vr < 8; vr++) {
        rowi[vr] = q0 + vr + 8 * half;
        padi[vr] = (ids[b * S_ + rowi[vr]] != PAD_ID) ? 1.f : 0.f;
    }

    float m[8], l[8];
    v8f O[4];
    #pragma unroll
    for (int vr = 0; vr < 8; vr++) { m[vr] = -1e30f; l[vr] = 0.f; }
    #pragma unroll
    for (int dg = 0; dg < 4; dg++)
        #pragma unroll
        for (int e = 0; e < 8; e++) O[dg][e] = 0.f;

    #pragma unroll 1
    for (int kt = 0; kt < S_; kt += 32) {
        // ---- scores for two 16-key subtiles ----
        v8f sc[2];
        int jcol[2];
        #pragma unroll
        for (int sub = 0; sub < 2; sub++) {
            jcol[sub] = kt + sub * 16 + lr;
            Frag16 bk0, bk1;
            #pragma unroll
            for (int j = 0; j < 8; j++) {
                size_t base = (size_t)(b * S_ + jcol[sub]) * H_ + hh * DH_;
                bk0.u[j] = kbu[(base + kkof(j, half)) >> 1];
                bk1.u[j] = kbu[(base + 32 + kkof(j, half)) >> 1];
            }
            v8f z;
            #pragma unroll
            for (int e = 0; e < 8; e++) z[e] = 0.f;
            z = wmma_bf16(qa[0].v, bk0.v, z);
            z = wmma_bf16(qa[1].v, bk1.v, z);
            sc[sub] = z;
        }

        // ---- mask + online softmax; stash P (bf16) into LDS ----
        int padj0 = (ids[b * S_ + jcol[0]] != PAD_ID);
        int padj1 = (ids[b * S_ + jcol[1]] != PAD_ID);
        #pragma unroll
        for (int vr = 0; vr < 8; vr++) {
            int i = rowi[vr];
            bool ans0 = (i >= qlv) && (jcol[0] >= qlv);
            bool ans1 = (i >= qlv) && (jcol[1] >= qlv);
            bool ok0 = (padi[vr] != 0.f) && padj0 && (!ans0 || jcol[0] <= i);
            bool ok1 = (padi[vr] != 0.f) && padj1 && (!ans1 || jcol[1] <= i);
            float x0 = sc[0][vr] * scale + (ok0 ? 0.f : -1e9f);
            float x1 = sc[1][vr] * scale + (ok1 ? 0.f : -1e9f);
            float rm = rowmax16(fmaxf(x0, x1));
            float mn = fmaxf(m[vr], rm);
            float al = __expf(m[vr] - mn);
            m[vr] = mn;
            float p0 = __expf(x0 - mn), p1 = __expf(x1 - mn);
            l[vr] = l[vr] * al + rowsum16(p0 + p1);
            #pragma unroll
            for (int dg = 0; dg < 4; dg++) O[dg][vr] *= al;
            Ps[(vr + 8 * half) * 40 + lr]      = f2bf(p0);
            Ps[(vr + 8 * half) * 40 + 16 + lr] = f2bf(p1);
        }
        __syncthreads();

        // ---- P (16x32) as A-fragment, V (32x64) as 4 B-fragments ----
        Frag16 pa;
        #pragma unroll
        for (int j = 0; j < 8; j++)
            pa.u[j] = *reinterpret_cast<const uint_t*>(&Ps[lr * 40 + kkof(j, half)]);
        #pragma unroll
        for (int dg = 0; dg < 4; dg++) {
            Frag16 bv;
            #pragma unroll
            for (int j = 0; j < 8; j++) {
                int kk = kkof(j, half);
                size_t r0 = (size_t)(b * S_ + kt + kk) * H_ + hh * DH_ + dg * 16 + lr;
                bv.s[2 * j]     = vb[r0];
                bv.s[2 * j + 1] = vb[r0 + H_];
            }
            O[dg] = wmma_bf16(pa.v, bv.v, O[dg]);
        }
        __syncthreads();
    }

    // ---- normalize + store context (bf16) ----
    #pragma unroll
    for (int dg = 0; dg < 4; dg++)
        #pragma unroll
        for (int vr = 0; vr < 8; vr++) {
            int row = q0 + vr + 8 * half;
            int col = hh * DH_ + dg * 16 + lr;
            float val = O[dg][vr] / l[vr];
            ctxb[(size_t)(b * S_ + row) * H_ + col] = f2bf(val);
        }
}

// ---------------------------------------------------------------------------
// Host-side orchestration
// ---------------------------------------------------------------------------
static size_t align256(size_t x) { return (x + 255) & ~(size_t)255; }

extern "C" void kernel_launch(void* const* d_in, const int* in_sizes, int n_in,
                              void* d_out, int out_size, void* d_ws, size_t ws_size,
                              hipStream_t stream) {
    (void)in_sizes; (void)n_in; (void)out_size; (void)ws_size;

    const int*   ids     = (const int*)  d_in[0];
    const float* wemb    = (const float*)d_in[1];
    const float* pemb    = (const float*)d_in[2];
    const float* temb    = (const float*)d_in[3];
    const float* eg      = (const float*)d_in[4];
    const float* eb      = (const float*)d_in[5];
    const float* attn_w  = (const float*)d_in[6];
    const float* attn_b  = (const float*)d_in[7];
    const float* ln1_g   = (const float*)d_in[8];
    const float* ln1_b   = (const float*)d_in[9];
    const float* ffn_w1  = (const float*)d_in[10];
    const float* ffn_b1  = (const float*)d_in[11];
    const float* ffn_w2  = (const float*)d_in[12];
    const float* ffn_b2  = (const float*)d_in[13];
    const float* ln2_g   = (const float*)d_in[14];
    const float* ln2_b   = (const float*)d_in[15];
    const float* cls_w   = (const float*)d_in[16];
    const float* cls_b   = (const float*)d_in[17];
    float* out = (float*)d_out;

    // ---- workspace carve-up ----
    char* base = (char*)d_ws;
    size_t cur = 0;
    auto take = [&](size_t bytes) { cur = align256(cur); void* p = base + cur; cur += bytes; return p; };

    float*    hF    = (float*)   take((size_t)M_ * H_ * 4);
    float*    tmpF  = (float*)   take((size_t)M_ * H_ * 4);
    ushort_t* hB    = (ushort_t*)take((size_t)M_ * H_ * 2);
    ushort_t* qB    = (ushort_t*)take((size_t)M_ * H_ * 2);
    ushort_t* kB    = (ushort_t*)take((size_t)M_ * H_ * 2);
    ushort_t* vB    = (ushort_t*)take((size_t)M_ * H_ * 2);
    ushort_t* ctxB  = (ushort_t*)take((size_t)M_ * H_ * 2);
    ushort_t* ffB   = (ushort_t*)take((size_t)M_ * FF_ * 2);
    ushort_t* wAtt  = (ushort_t*)take((size_t)L_ * 4 * H_ * H_ * 2);
    ushort_t* w1B   = (ushort_t*)take((size_t)L_ * H_ * FF_ * 2);
    ushort_t* w2B   = (ushort_t*)take((size_t)L_ * FF_ * H_ * 2);
    ushort_t* wCls  = (ushort_t*)take((size_t)H_ * V_ * 2);
    int*      qlenW = (int*)     take((size_t)B_ * 4);

    // ---- weight conversion to bf16 ----
    auto cvt = [&](const float* s, ushort_t* d, size_t n) {
        int grid = (int)((n + 256 * 8 - 1) / (256 * 8));
        cvt_bf16_kernel<<<grid, 256, 0, stream>>>(s, d, n);
    };
    cvt(attn_w, wAtt, (size_t)L_ * 4 * H_ * H_);
    cvt(ffn_w1, w1B,  (size_t)L_ * H_ * FF_);
    cvt(ffn_w2, w2B,  (size_t)L_ * FF_ * H_);
    cvt(cls_w,  wCls, (size_t)H_ * V_);

    qlen_kernel<<<B_, 32, 0, stream>>>(ids, qlenW);
    embed_ln_kernel<<<M_, 256, 0, stream>>>(ids, wemb, pemb, temb, eg, eb, hF, hB);

    const dim3 gH((H_ + 127) / 128, (M_ + 127) / 128);     // 6 x 32
    const dim3 gFF((FF_ + 127) / 128, (M_ + 127) / 128);   // 24 x 32
    const dim3 gV((V_ + 127) / 128, (M_ + 127) / 128);     // 166 x 32

    for (int l = 0; l < L_; l++) {
        const ushort_t* wq = wAtt + (size_t)l * 4 * H_ * H_;
        const ushort_t* wk = wq + (size_t)H_ * H_;
        const ushort_t* wv = wk + (size_t)H_ * H_;
        const ushort_t* wo = wv + (size_t)H_ * H_;
        const float* bq = attn_b + (size_t)l * 4 * H_;
        const float* bk = bq + H_;
        const float* bv = bk + H_;
        const float* bo = bv + H_;

        gemm_bf16_kernel<0><<<gH, 256, 0, stream>>>(hB, wq, bq, nullptr, qB, M_, H_, H_);
        gemm_bf16_kernel<0><<<gH, 256, 0, stream>>>(hB, wk, bk, nullptr, kB, M_, H_, H_);
        gemm_bf16_kernel<0><<<gH, 256, 0, stream>>>(hB, wv, bv, nullptr, vB, M_, H_, H_);

        flash_attn_kernel<<<B_ * NH_ * (S_ / 16), 32, 0, stream>>>(
            qB, kB, vB, ctxB, ids, qlenW);

        gemm_bf16_kernel<2><<<gH, 256, 0, stream>>>(ctxB, wo, bo, tmpF, nullptr, M_, H_, H_);
        add_ln_kernel<<<M_, 256, 0, stream>>>(hF, tmpF,
            ln1_g + (size_t)l * H_, ln1_b + (size_t)l * H_, hB);

        gemm_bf16_kernel<1><<<gFF, 256, 0, stream>>>(hB,
            w1B + (size_t)l * H_ * FF_, ffn_b1 + (size_t)l * FF_,
            nullptr, ffB, M_, FF_, H_);
        gemm_bf16_kernel<2><<<gH, 256, 0, stream>>>(ffB,
            w2B + (size_t)l * FF_ * H_, ffn_b2 + (size_t)l * H_,
            tmpF, nullptr, M_, H_, FF_);
        add_ln_kernel<<<M_, 256, 0, stream>>>(hF, tmpF,
            ln2_g + (size_t)l * H_, ln2_b + (size_t)l * H_, hB);
    }

    // classifier logits straight into d_out
    gemm_bf16_kernel<2><<<gV, 256, 0, stream>>>(hB, wCls, cls_b, out, nullptr, M_, V_, H_);
}
